// MultiHeadAttention_27986006901300
// MI455X (gfx1250) — matrix operand order repaired and checked by reading, not compiled, not run on hardware
//
#include <hip/hip_runtime.h>
#include <hip/hip_bf16.h>

// ---------------------------------------------------------------------------
// MHA forward for MI455X (gfx1250, wave32, WMMA 16x16x32 bf16, f32 accum).
//  B=2, S=2048, D=1024, H=16, DK=64.
//  Pipeline: f32->bf16 convert -> Q/K/V projections (WMMA GEMM) ->
//            flash attention (WMMA + TDM double-buffered K/V tiles in LDS,
//            online softmax) -> output projection.
//  Scratch layout in d_ws (needs 64 MiB):
//    [ 0MB) qb   bf16 [4096,1024]
//    [ 8MB) kb   bf16 [4096,1024]
//    [16MB) vb   bf16 [4096,1024]
//    [24MB) Wqb  bf16 [1024,1024]
//    [26MB) Wkb  bf16 [1024,1024]
//    [28MB) Wvb  bf16 [1024,1024]
//    [30MB) Wob  bf16 [1024,1024]
//    [32MB) Qh   bf16 [B,H,S,DK]
//    [40MB) Kh   bf16 [B,H,S,DK]
//    [48MB) Vt   bf16 [B,H,DK,S]   (transposed for PV A-operand locality)
//    [56MB) AO   bf16 [B,S,D]      (attention output, input to O-projection)
// ---------------------------------------------------------------------------

typedef __attribute__((ext_vector_type(16))) __bf16    v16bf;
typedef __attribute__((ext_vector_type(8)))  __bf16    bf16x8;
typedef __attribute__((ext_vector_type(8)))  float     v8f;
typedef __attribute__((ext_vector_type(4)))  unsigned  v4u;
typedef __attribute__((ext_vector_type(8)))  int       v8i;
typedef __attribute__((ext_vector_type(4)))  int       v4i;

#define WMMA_BF16(a, b, c) \
  __builtin_amdgcn_wmma_f32_16x16x32_bf16(false, (a), false, (b), (short)0, (c), false, false)

constexpr int Bsz = 2, S = 2048, D = 1024, H = 16, DK = 64;

__device__ inline unsigned pk2bf(float a, float b) {
  union { __bf16 h[2]; unsigned u; } t;
  t.h[0] = (__bf16)a; t.h[1] = (__bf16)b;
  return t.u;
}

// ---------------------------------------------------------------------------
// TDM: 2-D tile load Global -> LDS via Tensor Data Mover (D# per ISA ch.8).
//  group0: [1:0]=count=1, [63:32]=lds byte addr, [120:64]=global byte addr,
//          [127:126]=type=2
//  group1: [17:16]=data_size(1 => 2 bytes), [79:48]=tensor_dim0,
//          [111:80]=tensor_dim1, [127:112]=tile_dim0, [143:128]=tile_dim1,
//          [207:160]=tensor_dim0_stride   (dims/strides in elements)
//  groups 2/3 + trailing group: zero (<=2D tensor)
//  This toolchain exposes the 6-operand builtin:
//    (uint32x4, int32x8, int32x4, int32x4, int32x8, i32 cpol)
// ---------------------------------------------------------------------------
__device__ inline void tdm_load_2d(unsigned lds_off, unsigned long long gaddr,
                                   unsigned tile_d0, unsigned tile_d1,
                                   unsigned tens_d0, unsigned tens_d1,
                                   unsigned stride0) {
  v4u g0 = { 1u,                                   // count=1, user descriptor
             lds_off,                              // LDS byte address
             (unsigned)gaddr,                      // global addr [31:0]
             (unsigned)((gaddr >> 32) & 0x01ffffffu) | (2u << 30) }; // [56:32]+type
  v8i g1 = { (int)(1u << 16),                      // data_size = 2 bytes
             (int)(tens_d0 << 16),                 // tensor_dim0[15:0]
             (int)((tens_d0 >> 16) | (tens_d1 << 16)),
             (int)(((tens_d1 >> 16) & 0xffffu) | (tile_d0 << 16)),
             (int)tile_d1,                         // tile_dim1 (tile_dim2=0)
             (int)stride0,                         // tensor_dim0_stride[31:0]
             0, 0 };
  v4i gz4 = { 0, 0, 0, 0 };
  v8i gz8 = { 0, 0, 0, 0, 0, 0, 0, 0 };
  __builtin_amdgcn_tensor_load_to_lds(g0, g1, gz4, gz4, gz8, 0);
}

// ---------------------------------------------------------------------------
// Elementwise f32 -> bf16 (RNE via __bf16 cast)
// ---------------------------------------------------------------------------
__global__ __launch_bounds__(256) void cvt_f32_bf16(const float* __restrict__ s,
                                                    __bf16* __restrict__ d, int n) {
  for (int i = blockIdx.x * blockDim.x + threadIdx.x; i < n; i += gridDim.x * blockDim.x)
    d[i] = (__bf16)s[i];
}

// ---------------------------------------------------------------------------
// GEMM: Y[m,n] = sum_k X[m,k]*W[n,k] + bias[n]   (torch Linear, W row-major [N,K])
// One wave computes a 16(M) x 64(N) tile; K loop in steps of 32 -> 4 WMMAs/step.
// mode: 0 Q->[B,H,S,DK] bf16, 1 K->same, 2 V->[B,H,DK,S] bf16, 3 O->f32 [M,N]
// ---------------------------------------------------------------------------
__global__ __launch_bounds__(256) void proj_gemm(const __bf16* __restrict__ X,
                                                 const __bf16* __restrict__ W,
                                                 const float* __restrict__ bias,
                                                 __bf16* __restrict__ Yb,
                                                 float* __restrict__ Yf, int mode) {
  const int lane = threadIdx.x & 31;
  const int wave = threadIdx.x >> 5;
  const int gw   = blockIdx.x * 8 + wave;
  const int NG   = D / 64;                    // 16 n-groups
  const int m0   = (gw / NG) * 16;
  const int n0   = (gw % NG) * 64;
  if (m0 >= Bsz * S) return;

  const int lr = lane & 15;
  const int hi = lane >> 4;                   // 0/1: K-half select
  const __bf16* Xrow = X + (size_t)(m0 + lr) * D;
  const int kShift = hi * 16;

  v8f acc0 = {}, acc1 = {}, acc2 = {}, acc3 = {};
  for (int k0 = 0; k0 < D; k0 += 32) {
    union { v16bf v; bf16x8 h[2]; } A;
    A.h[0] = *(const bf16x8*)(Xrow + k0 + kShift);
    A.h[1] = *(const bf16x8*)(Xrow + k0 + 8 + kShift);
    const int kc = k0 + hi * 16;
    const __bf16* W0 = W + (size_t)(n0 + lr) * D + kc;
    v16bf b0 = *(const v16bf*)(W0);
    v16bf b1 = *(const v16bf*)(W0 + 16 * (size_t)D);
    v16bf b2 = *(const v16bf*)(W0 + 32 * (size_t)D);
    v16bf b3 = *(const v16bf*)(W0 + 48 * (size_t)D);
    acc0 = WMMA_BF16(A.v, b0, acc0);
    acc1 = WMMA_BF16(A.v, b1, acc1);
    acc2 = WMMA_BF16(A.v, b2, acc2);
    acc3 = WMMA_BF16(A.v, b3, acc3);
  }

  // C layout: VGPR r, lane l -> M = r + (l<16?0:8), N = l%16
  v8f accs[4] = {acc0, acc1, acc2, acc3};
  const int rb = hi * 8;
#pragma unroll
  for (int t = 0; t < 4; t++) {
    const int n = n0 + t * 16 + lr;
    const float bv = bias[n];
#pragma unroll
    for (int r = 0; r < 8; r++) {
      const float y = accs[t][r] + bv;
      const int m = m0 + r + rb;
      if (mode == 3) {
        Yf[(size_t)m * D + n] = y;
      } else {
        const int b = m >> 11, s = m & (S - 1);
        const int h = n >> 6,  dk = n & 63;
        size_t idx;
        if (mode == 2) idx = (((size_t)(b * H + h)) * DK + dk) * S + s;   // V transposed
        else           idx = (((size_t)(b * H + h)) * S + s) * DK + dk;   // Q/K head-major
        Yb[idx] = (__bf16)y;
      }
    }
  }
}

// ---------------------------------------------------------------------------
// Flash attention. One wave per 16-query tile; the 8 waves of a block share
// one (b,h), so K/V 32-key tiles are staged ONCE per block into LDS by the
// Tensor Data Mover (double-buffered), synchronized with s_wait_tensorcnt +
// workgroup barriers. S^T = K*Q^T so per-query softmax stats need only one
// shfl_xor(16) combine; O^T = V^T*P rescales per-lane.
// Dynamic LDS (16 KiB): kbuf[2][32][64] bf16 @ 0/4096, vbuf[2][64][32] @ 8192/12288.
// ---------------------------------------------------------------------------
__global__ __launch_bounds__(256) void flash_attn(const __bf16* __restrict__ Qh,
                                                  const __bf16* __restrict__ Kh,
                                                  const __bf16* __restrict__ Vt,
                                                  const unsigned char* __restrict__ mask,
                                                  __bf16* __restrict__ AO) {
  extern __shared__ __bf16 smem[];   // dynamic LDS only -> offsets start at 0

  const int lane = threadIdx.x & 31;
  const int wave = threadIdx.x >> 5;
  const int gw   = blockIdx.x * 8 + wave;
  const int QT   = S / 16;                     // 128 query tiles per head
  const int b    = gw / (H * QT);
  const int h    = (gw / QT) % H;
  const int q0   = (gw % QT) * 16;

  const int lr = lane & 15, hi = lane >> 4;
  const int rb = hi * 8, ksh = hi * 16;
  const __bf16* Qb = Qh + ((size_t)(b * H + h)) * S * DK;
  const __bf16* Kb = Kh + ((size_t)(b * H + h)) * S * DK;
  const __bf16* Vb = Vt + ((size_t)(b * H + h)) * DK * S;
  const unsigned char* mrow = mask + ((size_t)b * S + (q0 + lr)) * S;

  const unsigned long long kgbase = (unsigned long long)(uintptr_t)Kb;
  const unsigned long long vgbase = (unsigned long long)(uintptr_t)Vb;

  // Q^T B-operands: lane = query col, K-dim = d (two 32-d chunks)
  const __bf16* Qrow = Qb + (size_t)(q0 + lr) * DK + hi * 16;
  const v16bf qb0 = *(const v16bf*)(Qrow);        // d 0..31
  const v16bf qb1 = *(const v16bf*)(Qrow + 32);   // d 32..63

  v8f o0 = {}, o1 = {}, o2 = {}, o3 = {};        // O^T tiles, M = d 0..63
  float mrun = -1e30f, lrun = 0.0f;

  const int NIT = S / 32;                        // 64 key-tile iterations

  // prologue: stage tiles for iteration 0
  if (wave == 0) {
    // K tile: 32 key-rows x 64 d (row-contiguous, stride DK)
    tdm_load_2d(0u, kgbase, /*tile*/ 64, 32, /*tensor*/ 64, S, /*stride0*/ DK);
    // V tile: 64 d-rows x 32 keys (row stride S)
    tdm_load_2d(8192u, vgbase, /*tile*/ 32, 64, /*tensor*/ S, DK, /*stride0*/ S);
  }

  for (int it = 0; it < NIT; ++it) {
    const int kt = it * 32;
    const int p  = it & 1;

    if (wave == 0) {
      if (it + 1 < NIT) {
        const int kn = (it + 1) * 32;
        const unsigned pn = (unsigned)((it + 1) & 1);
        tdm_load_2d(pn * 4096u, kgbase + (unsigned long long)kn * DK * 2, 64, 32, 64, S, DK);
        tdm_load_2d(8192u + pn * 4096u, vgbase + (unsigned long long)kn * 2, 32, 64, S, DK, S);
        __builtin_amdgcn_s_wait_tensorcnt(2);   // iter-it pair done (TDM in-order)
      } else {
        __builtin_amdgcn_s_wait_tensorcnt(0);
      }
    }
    __syncthreads();                            // LDS tiles visible to all waves

    const __bf16* kb_l = smem + (unsigned)p * 2048u;           // [32][64]
    const __bf16* vb_l = smem + 4096u + (unsigned)p * 2048u;   // [64][32]

    // K A-operands from LDS: rows = keys, K-dim = d
    const __bf16* K0 = kb_l + lr * DK;
    const __bf16* K1 = kb_l + (16 + lr) * DK;
    union { v16bf v; bf16x8 hh[2]; } a00, a01, a10, a11;
    a00.hh[0] = *(const bf16x8*)(K0 + ksh);      a00.hh[1] = *(const bf16x8*)(K0 + 8 + ksh);
    a01.hh[0] = *(const bf16x8*)(K0 + 32 + ksh); a01.hh[1] = *(const bf16x8*)(K0 + 40 + ksh);
    a10.hh[0] = *(const bf16x8*)(K1 + ksh);      a10.hh[1] = *(const bf16x8*)(K1 + 8 + ksh);
    a11.hh[0] = *(const bf16x8*)(K1 + 32 + ksh); a11.hh[1] = *(const bf16x8*)(K1 + 40 + ksh);

    v8f s0 = {}, s1 = {};
    s0 = WMMA_BF16(a00.v, qb0, s0);
    s0 = WMMA_BF16(a01.v, qb1, s0);
    s1 = WMMA_BF16(a10.v, qb0, s1);
    s1 = WMMA_BF16(a11.v, qb1, s1);

    // scale + mask; element (r,lane): key = kt + tile*16 + r + rb, query = q0+lr
    float tmax = -1e30f;
#pragma unroll
    for (int r = 0; r < 8; r++) {
      float e0 = s0[r] * 0.125f; if (!mrow[kt + r + rb])      e0 = -1e9f;
      float e1 = s1[r] * 0.125f; if (!mrow[kt + 16 + r + rb]) e1 = -1e9f;
      s0[r] = e0; s1[r] = e1;
      tmax = fmaxf(tmax, fmaxf(e0, e1));
    }
    tmax = fmaxf(tmax, __shfl_xor(tmax, 16, 32));
    const float mnew = fmaxf(mrun, tmax);
    const float scl  = __expf(mrun - mnew);
    float tsum = 0.0f;
#pragma unroll
    for (int r = 0; r < 8; r++) {
      const float p0 = __expf(s0[r] - mnew);
      const float p1 = __expf(s1[r] - mnew);
      s0[r] = p0; s1[r] = p1;
      tsum += p0 + p1;
    }
    tsum += __shfl_xor(tsum, 16, 32);
    lrun = lrun * scl + tsum;
    mrun = mnew;
#pragma unroll
    for (int r = 0; r < 8; r++) { o0[r] *= scl; o1[r] *= scl; o2[r] *= scl; o3[r] *= scl; }

    // P (keys x queries) C-layout -> B-operand layout via half-wave swaps
    union { v16bf v; unsigned u[8]; } P;
#pragma unroll
    for (int j = 0; j < 4; j++) {
      const unsigned p0 = pk2bf(s0[2 * j], s0[2 * j + 1]);
      const unsigned p1 = pk2bf(s1[2 * j], s1[2 * j + 1]);
      const unsigned x0 = __shfl_xor(p0, 16, 32);
      const unsigned x1 = __shfl_xor(p1, 16, 32);
      P.u[j]     = (lane < 16) ? p0 : x1;
      P.u[j + 4] = (lane < 16) ? x0 : p1;
    }

    // O^T += V^T * P  (4 d-tiles; A rows = d, keys contiguous in LDS vbuf)
    union { v16bf v; bf16x8 hh[2]; } va;
    const __bf16* V0 = vb_l + (0 * 16 + lr) * 32 + ksh;
    va.hh[0] = *(const bf16x8*)(V0); va.hh[1] = *(const bf16x8*)(V0 + 8);
    o0 = WMMA_BF16(va.v, P.v, o0);
    const __bf16* V1 = vb_l + (1 * 16 + lr) * 32 + ksh;
    va.hh[0] = *(const bf16x8*)(V1); va.hh[1] = *(const bf16x8*)(V1 + 8);
    o1 = WMMA_BF16(va.v, P.v, o1);
    const __bf16* V2 = vb_l + (2 * 16 + lr) * 32 + ksh;
    va.hh[0] = *(const bf16x8*)(V2); va.hh[1] = *(const bf16x8*)(V2 + 8);
    o2 = WMMA_BF16(va.v, P.v, o2);
    const __bf16* V3 = vb_l + (3 * 16 + lr) * 32 + ksh;
    va.hh[0] = *(const bf16x8*)(V3); va.hh[1] = *(const bf16x8*)(V3 + 8);
    o3 = WMMA_BF16(va.v, P.v, o3);

    __syncthreads();                            // all reads done before rewrite
  }

  // epilogue: O^T lane = query col; VGPR r of tile dt -> d = dt*16 + r + rb
  const float inv = 1.0f / lrun;
  __bf16* orow = AO + ((size_t)b * S + (q0 + lr)) * D + h * DK;
#pragma unroll
  for (int r = 0; r < 8; r++) {
    orow[0 * 16 + r + rb] = (__bf16)(o0[r] * inv);
    orow[1 * 16 + r + rb] = (__bf16)(o1[r] * inv);
    orow[2 * 16 + r + rb] = (__bf16)(o2[r] * inv);
    orow[3 * 16 + r + rb] = (__bf16)(o3[r] * inv);
  }
}

// ---------------------------------------------------------------------------
extern "C" void kernel_launch(void* const* d_in, const int* in_sizes, int n_in,
                              void* d_out, int out_size, void* d_ws, size_t ws_size,
                              hipStream_t stream) {
  (void)in_sizes; (void)n_in; (void)out_size; (void)ws_size;
  const float* q  = (const float*)d_in[0];
  const float* k  = (const float*)d_in[1];
  const float* v  = (const float*)d_in[2];
  const unsigned char* mask = (const unsigned char*)d_in[3];
  const float* Wq = (const float*)d_in[4];
  const float* bq = (const float*)d_in[5];
  const float* Wk = (const float*)d_in[6];
  const float* bk = (const float*)d_in[7];
  const float* Wv = (const float*)d_in[8];
  const float* bv = (const float*)d_in[9];
  const float* Wo = (const float*)d_in[10];
  const float* bo = (const float*)d_in[11];
  float* out = (float*)d_out;

  char* ws = (char*)d_ws;
  const size_t MB = 1u << 20;
  __bf16* qb  = (__bf16*)(ws + 0 * MB);
  __bf16* kb  = (__bf16*)(ws + 8 * MB);
  __bf16* vb  = (__bf16*)(ws + 16 * MB);
  __bf16* Wqb = (__bf16*)(ws + 24 * MB);
  __bf16* Wkb = (__bf16*)(ws + 26 * MB);
  __bf16* Wvb = (__bf16*)(ws + 28 * MB);
  __bf16* Wob = (__bf16*)(ws + 30 * MB);
  __bf16* Qhh = (__bf16*)(ws + 32 * MB);
  __bf16* Khh = (__bf16*)(ws + 40 * MB);
  __bf16* Vtt = (__bf16*)(ws + 48 * MB);
  __bf16* AO  = (__bf16*)(ws + 56 * MB);

  const int NTOK = Bsz * S * D;   // 4M elements
  const int NW   = D * D;         // 1M elements
  cvt_f32_bf16<<<1024, 256, 0, stream>>>(q,  qb,  NTOK);
  cvt_f32_bf16<<<1024, 256, 0, stream>>>(k,  kb,  NTOK);
  cvt_f32_bf16<<<1024, 256, 0, stream>>>(v,  vb,  NTOK);
  cvt_f32_bf16<<<512,  256, 0, stream>>>(Wq, Wqb, NW);
  cvt_f32_bf16<<<512,  256, 0, stream>>>(Wk, Wkb, NW);
  cvt_f32_bf16<<<512,  256, 0, stream>>>(Wv, Wvb, NW);
  cvt_f32_bf16<<<512,  256, 0, stream>>>(Wo, Wob, NW);

  // (B*S/16) m-tiles * (D/64) n-groups = 4096 waves = 512 blocks of 8 waves
  proj_gemm<<<512, 256, 0, stream>>>(qb, Wqb, bq, Qhh, nullptr, 0);
  proj_gemm<<<512, 256, 0, stream>>>(kb, Wkb, bk, Khh, nullptr, 1);
  proj_gemm<<<512, 256, 0, stream>>>(vb, Wvb, bv, Vtt, nullptr, 2);

  // B*H*(S/16) = 4096 waves = 512 blocks; 16 KiB dynamic LDS for TDM buffers
  flash_attn<<<512, 256, 16384, stream>>>(Qhh, Khh, Vtt, mask, AO);

  proj_gemm<<<512, 256, 0, stream>>>(AO, Wob, bo, nullptr, out, 3);
}
